// GCN_61306363183369
// MI455X (gfx1250) — compile-verified, matrix operand order checked by reading, not run on hardware
//
#include <hip/hip_runtime.h>
#include <hip/hip_bf16.h>

// ---- WMMA vector types (gfx1250, wave32) ----
typedef __attribute__((ext_vector_type(16))) __bf16 v16bf;
typedef __attribute__((ext_vector_type(8)))  __bf16 v8bf;
typedef __attribute__((ext_vector_type(8)))  float  v8f;

#define IN_F  256
#define H_F   256
#define OUT_F 128

// ---------------- utility kernels ----------------

__global__ void zero_f32_kernel(float* __restrict__ p, long long n) {
    long long t = (long long)blockIdx.x * blockDim.x + threadIdx.x;
    if (t < n) p[t] = 0.0f;
}

__global__ void degree_kernel(const int* __restrict__ src, const int* __restrict__ dst,
                              float* __restrict__ degO, float* __restrict__ degI, int E) {
    int e = blockIdx.x * blockDim.x + threadIdx.x;
    if (e < E) {
        atomicAdd(&degO[src[e]], 1.0f);
        atomicAdd(&degI[dst[e]], 1.0f);
    }
}

__global__ void f32_to_bf16_kernel(const float* __restrict__ x, __bf16* __restrict__ y,
                                   long long n) {
    long long t = (long long)blockIdx.x * blockDim.x + threadIdx.x;
    if (t < n) y[t] = (__bf16)x[t];
}

// Repack W [K x N] (row-major f32) into per-wave B-fragment layout:
// fragment (ntile, kb) = 32 lanes x 16 bf16, lane-major contiguous (32B per lane).
// Lane l holds column  n = ntile*16 + (l & 15); with kg = (l>>4)*8:
//   elems 0..7  -> K = kb*32 + kg + i
//   elems 8..15 -> K = kb*32 + 16 + kg + (i-8)
__global__ void pack_w_kernel(const float* __restrict__ W, __bf16* __restrict__ frag,
                              int K, int N) {
    int idx = blockIdx.x * blockDim.x + threadIdx.x;
    int kblocks = K >> 5;
    int total = (N >> 4) * kblocks * 512;
    if (idx >= total) return;
    int i     = idx & 15;
    int lane  = (idx >> 4) & 31;
    int rest  = idx >> 9;
    int kb    = rest % kblocks;
    int ntile = rest / kblocks;
    int col   = ntile * 16 + (lane & 15);
    int kg    = (lane >> 4) * 8;
    int k     = kb * 32 + ((i < 8) ? (kg + i) : (16 + kg + (i - 8)));
    frag[idx] = (__bf16)W[(size_t)k * N + col];
}

// ---------------- WMMA GEMM: C[M x N] = (A_bf16[M x K] @ B_frag) * norm_out(row) ----------------
// One 16x16 output tile per wave, 4 waves per block (blockDim.x = 128).
// grid.x = ceil(M/16), grid.y = N/64. K templated so the 8-step K loop fully unrolls
// into 8 v_wmma_f32_16x16x32_bf16 instructions.
template <int K>
__global__ __launch_bounds__(128) void wmma_gemm_rowscale_kernel(
    const __bf16* __restrict__ A, const __bf16* __restrict__ Bfrag,
    const float* __restrict__ degO, float* __restrict__ C, int M, int N) {
    constexpr int KB = K / 32;
    const int lane  = threadIdx.x & 31;
    const int wave  = threadIdx.x >> 5;
    const int mtile = blockIdx.x;
    const int ntile = blockIdx.y * 4 + wave;   // wave-uniform

    // A-fragment addressing (ISA 7.12.2, 16-bit A 16x32):
    const int ra = lane & 15;          // row within M-tile
    const int kg = (lane >> 4) * 8;    // K sub-group: lanes 0-15 -> 0, 16-31 -> 8
    int arow = mtile * 16 + ra;
    if (arow > M - 1) arow = M - 1;    // clamp (uniform flow; EXEC stays all-1s)
    const __bf16* Ap = A + (size_t)arow * K;
    const __bf16* Bp = Bfrag + (size_t)ntile * KB * 512;

    v8f acc = {};
#pragma unroll
    for (int kb = 0; kb < KB; ++kb) {
        v8bf lo = *(const v8bf*)(Ap + kb * 32 + kg);        // K = kb*32 + kg + [0..7]
        v8bf hi = *(const v8bf*)(Ap + kb * 32 + 16 + kg);   // K = kb*32 + 16 + kg + [0..7]
        v16bf a;
#pragma unroll
        for (int i = 0; i < 8; ++i) { a[i] = lo[i]; a[8 + i] = hi[i]; }
        v16bf b = *(const v16bf*)(Bp + ((size_t)kb * 32 + lane) * 16);
        // Prefetch next B fragment (speculative; OOB prefetch is dropped by HW).
        __builtin_prefetch((const void*)(Bp + ((size_t)(kb + 1) * 32 + lane) * 16), 0, 1);
        acc = __builtin_amdgcn_wmma_f32_16x16x32_bf16(
            /*neg_a=*/false, a, /*neg_b=*/false, b,
            /*c_mod=*/(short)0, acc, /*reuse_a=*/false, /*reuse_b=*/false);
    }

    // C/D layout: VGPR r, lanes 0-15 -> M = r, lanes 16-31 -> M = 8 + r; N = lane & 15.
    const int col   = ntile * 16 + (lane & 15);
    const int rbase = mtile * 16 + (lane >> 4) * 8;
#pragma unroll
    for (int r = 0; r < 8; ++r) {
        int row = rbase + r;
        if (row < M) {
            float s = rsqrtf(fmaxf(degO[row], 1.0f));
            C[(size_t)row * N + col] = acc[r] * s;
        }
    }
}

// ---------------- SpMM scatter: AGG[dst] += H[src] ----------------
// grid.x covers edges, grid.y covers feature quads; float4 gather + 4 atomic f32 adds.
__global__ void spmm_scatter_kernel(const int* __restrict__ src, const int* __restrict__ dst,
                                    const float* __restrict__ H, float* __restrict__ AGG,
                                    int E, int F) {
    int e = blockIdx.x * blockDim.x + threadIdx.x;
    if (e >= E) return;
    int g = blockIdx.y * 4;
    int s = src[e], d = dst[e];
    float4 v = *(const float4*)(H + (size_t)s * F + g);
    float* ap = AGG + (size_t)d * F + g;
    atomicAdd(ap + 0, v.x);
    atomicAdd(ap + 1, v.y);
    atomicAdd(ap + 2, v.z);
    atomicAdd(ap + 3, v.w);
}

// ---------------- post-op: h = relu(agg * norm_in + b) -> bf16 ----------------
__global__ void post_relu_bf16_kernel(const float* __restrict__ agg,
                                      const float* __restrict__ degI,
                                      const float* __restrict__ b,
                                      __bf16* __restrict__ out, int Nn, int F) {
    long long t = (long long)blockIdx.x * blockDim.x + threadIdx.x;
    long long total = (long long)Nn * F;
    if (t >= total) return;
    int n = (int)(t / F);
    int f = (int)(t % F);
    float s = rsqrtf(fmaxf(degI[n], 1.0f));
    float v = agg[t] * s + b[f];
    out[t] = (__bf16)fmaxf(v, 0.0f);
}

// ---------------- final: out[f] = mean_n(agg2[n,f]*norm_in[n]) + b2[f] ----------------
__global__ void mean_reduce_kernel(const float* __restrict__ agg2,
                                   const float* __restrict__ degI,
                                   const float* __restrict__ b2,
                                   float* __restrict__ out, int Nn, int F) {
    int f = blockIdx.x;
    float sum = 0.0f;
    for (int n = threadIdx.x; n < Nn; n += blockDim.x) {
        float s = rsqrtf(fmaxf(degI[n], 1.0f));
        sum += agg2[(size_t)n * F + f] * s;
    }
    __shared__ float red[256];
    red[threadIdx.x] = sum;
    __syncthreads();
    for (int off = 128; off > 0; off >>= 1) {
        if (threadIdx.x < off) red[threadIdx.x] += red[threadIdx.x + off];
        __syncthreads();
    }
    if (threadIdx.x == 0) out[f] = red[0] / (float)Nn + b2[f];
}

// ---------------- host launcher ----------------
extern "C" void kernel_launch(void* const* d_in, const int* in_sizes, int n_in,
                              void* d_out, int out_size, void* d_ws, size_t ws_size,
                              hipStream_t stream) {
    const float* in_feat = (const float*)d_in[0];
    const int*   src     = (const int*)d_in[1];
    const int*   dst     = (const int*)d_in[2];
    const float* W1      = (const float*)d_in[3];
    const float* b1      = (const float*)d_in[4];
    const float* W2      = (const float*)d_in[5];
    const float* b2      = (const float*)d_in[6];
    float* out = (float*)d_out;

    const int Nn = in_sizes[0] / IN_F;   // 50000
    const int E  = in_sizes[1];          // 800000
    (void)n_in; (void)out_size; (void)ws_size;

    // Workspace carve-out (256B aligned); Hpre/AGG reused across layers.
    char* ws = (char*)d_ws;
    size_t off = 0;
    auto carve = [&](size_t bytes) -> char* {
        char* p = ws + off;
        off = (off + bytes + 255) & ~(size_t)255;
        return p;
    };
    __bf16* Xbf  = (__bf16*)carve((size_t)Nn * IN_F * 2);
    __bf16* W1f  = (__bf16*)carve((size_t)(H_F / 16) * (IN_F / 32) * 512 * 2);
    __bf16* W2f  = (__bf16*)carve((size_t)(OUT_F / 16) * (H_F / 32) * 512 * 2);
    float*  degO = (float*)carve((size_t)Nn * 4);
    float*  degI = (float*)carve((size_t)Nn * 4);
    float*  Hpre = (float*)carve((size_t)Nn * H_F * 4);   // layer1 H, reused for layer2
    float*  AGG  = (float*)carve((size_t)Nn * H_F * 4);   // layer1 agg, reused for layer2
    __bf16* H1bf = (__bf16*)carve((size_t)Nn * H_F * 2);

    const int TB = 256;
    auto blocks = [&](long long n) { return (unsigned)((n + TB - 1) / TB); };

    // 1) degrees
    zero_f32_kernel<<<blocks(Nn), TB, 0, stream>>>(degO, Nn);
    zero_f32_kernel<<<blocks(Nn), TB, 0, stream>>>(degI, Nn);
    degree_kernel<<<blocks(E), TB, 0, stream>>>(src, dst, degO, degI, E);

    // 2) precision conversion + weight fragment packing
    f32_to_bf16_kernel<<<blocks((long long)Nn * IN_F), TB, 0, stream>>>(
        in_feat, Xbf, (long long)Nn * IN_F);
    pack_w_kernel<<<blocks((H_F / 16) * (IN_F / 32) * 512), TB, 0, stream>>>(W1, W1f, IN_F, H_F);
    pack_w_kernel<<<blocks((OUT_F / 16) * (H_F / 32) * 512), TB, 0, stream>>>(W2, W2f, H_F, OUT_F);

    // 3) layer 1: GEMM (WMMA bf16) with fused norm_out row-scale
    dim3 g1((Nn + 15) / 16, H_F / 64);
    wmma_gemm_rowscale_kernel<IN_F><<<g1, 128, 0, stream>>>(Xbf, W1f, degO, Hpre, Nn, H_F);

    // 4) layer 1: SpMM scatter-add
    zero_f32_kernel<<<blocks((long long)Nn * H_F), TB, 0, stream>>>(AGG, (long long)Nn * H_F);
    dim3 gs1(blocks(E), H_F / 4);
    spmm_scatter_kernel<<<gs1, TB, 0, stream>>>(src, dst, Hpre, AGG, E, H_F);

    // 5) norm_in scale + bias + ReLU -> bf16 activations
    post_relu_bf16_kernel<<<blocks((long long)Nn * H_F), TB, 0, stream>>>(
        AGG, degI, b1, H1bf, Nn, H_F);

    // 6) layer 2: GEMM (WMMA bf16) with fused norm_out row-scale
    dim3 g2((Nn + 15) / 16, OUT_F / 64);
    wmma_gemm_rowscale_kernel<H_F><<<g2, 128, 0, stream>>>(H1bf, W2f, degO, Hpre, Nn, OUT_F);

    // 7) layer 2: SpMM scatter-add
    zero_f32_kernel<<<blocks((long long)Nn * OUT_F), TB, 0, stream>>>(AGG, (long long)Nn * OUT_F);
    dim3 gs2(blocks(E), OUT_F / 4);
    spmm_scatter_kernel<<<gs2, TB, 0, stream>>>(src, dst, Hpre, AGG, E, OUT_F);

    // 8) norm_in scale + bias + mean over nodes -> out[OUT_F]
    mean_reduce_kernel<<<OUT_F, 256, 0, stream>>>(AGG, degI, b2, out, Nn, OUT_F);
}
